// Gemma4RoutedExperts_28802050687684
// MI455X (gfx1250) — compile-verified, matrix operand order checked by reading, not run on hardware
//
#include <hip/hip_runtime.h>
#include <hip/hip_bf16.h>
#include <math.h>

typedef __attribute__((ext_vector_type(16))) __bf16 v16bf;
typedef __attribute__((ext_vector_type(8)))  float  v8f;

#define NUM_EXPERTS 8
#define EMBED       1024
#define EXPERT      2048
#define TOPK        2
#define NTOK        2048            // B*T = 2*1024
#define NASSIGN     (NTOK * TOPK)   // 4096
#define TILE_M      32
#define MAX_TILES   (NASSIGN / TILE_M)  // 128 (worst case: all tokens on one expert)
#define CHUNK       512
#define NCHUNK      (EXPERT / CHUNK)    // 4
#define XS_STRIDE   (EMBED + 8)     // +8 bf16 = 16B pad: rows land 4 dwords apart -> no bank conflicts
#define HS_STRIDE   (CHUNK + 8)

__device__ __forceinline__ float fast_sigmoid(float s) {
    // 1/(1+e^-s) with v_exp_f32 + v_rcp_f32 (no IEEE div macro)
    return __builtin_amdgcn_rcpf(1.0f + __expf(-s));
}

// ---------------------------------------------------------------------------
// 0) zero output + routing counters
// ---------------------------------------------------------------------------
__global__ void zero_kernel(float* __restrict__ out, int* __restrict__ counts) {
    int i = blockIdx.x * 256 + threadIdx.x;
    if (i < NTOK * EMBED) out[i] = 0.0f;
    if (i < NUM_EXPERTS)  counts[i] = 0;
}

// ---------------------------------------------------------------------------
// 1) gating logits: [2048,1024] x [1024,8] -> [2048,8]
// ---------------------------------------------------------------------------
__global__ void gate_logits_kernel(const float* __restrict__ g,
                                   const float* __restrict__ gk,
                                   float* __restrict__ logits) {
    int idx = blockIdx.x * 256 + threadIdx.x;   // t*8 + e
    if (idx >= NTOK * NUM_EXPERTS) return;
    int t = idx >> 3;
    int e = idx & 7;
    const float* row = g + (size_t)t * EMBED;
    float s = 0.0f;
    for (int i = 0; i < EMBED; ++i)
        s += row[i] * gk[i * NUM_EXPERTS + e];
    logits[idx] = s;
}

// ---------------------------------------------------------------------------
// 2) per-token top-2 + sigmoid weights + expert histogram
// ---------------------------------------------------------------------------
__global__ void topk_kernel(const float* __restrict__ logits,
                            int* __restrict__ topk_idx,
                            float* __restrict__ topk_w,
                            int* __restrict__ counts) {
    int t = blockIdx.x * 256 + threadIdx.x;
    if (t >= NTOK) return;
    const float* l = logits + (size_t)t * NUM_EXPERTS;
    float v0 = -1e30f, v1 = -1e30f;
    int   b0 = 0,      b1 = 0;
    for (int e = 0; e < NUM_EXPERTS; ++e) {
        float v = l[e];
        if (v > v0)      { v1 = v0; b1 = b0; v0 = v; b0 = e; }
        else if (v > v1) { v1 = v;  b1 = e; }
    }
    topk_idx[t * 2 + 0] = b0;
    topk_idx[t * 2 + 1] = b1;
    topk_w[t * 2 + 0] = fast_sigmoid(v0);
    topk_w[t * 2 + 1] = fast_sigmoid(v1);
    atomicAdd(&counts[b0], 1);
    atomicAdd(&counts[b1], 1);
}

// ---------------------------------------------------------------------------
// 3) exclusive scan of the 8 counts + reset cursors
// ---------------------------------------------------------------------------
__global__ void scan_kernel(const int* __restrict__ counts,
                            int* __restrict__ offsets,
                            int* __restrict__ cursors) {
    if (threadIdx.x == 0) {
        int s = 0;
        for (int e = 0; e < NUM_EXPERTS; ++e) { offsets[e] = s; s += counts[e]; }
    }
    if (threadIdx.x < NUM_EXPERTS) cursors[threadIdx.x] = 0;
}

// ---------------------------------------------------------------------------
// 4) scatter assignment ids into per-expert buckets
// ---------------------------------------------------------------------------
__global__ void scatter_kernel(const int* __restrict__ topk_idx,
                               const int* __restrict__ offsets,
                               int* __restrict__ cursors,
                               int* __restrict__ bucket) {
    int a = blockIdx.x * 256 + threadIdx.x;
    if (a >= NASSIGN) return;
    int e = topk_idx[a];
    int p = atomicAdd(&cursors[e], 1);
    bucket[offsets[e] + p] = a;
}

// ---------------------------------------------------------------------------
// 5) grouped expert GEMM tile kernel (bf16 WMMA, fp32 accumulate)
//    block = 256 threads = 8 wave32; one 32-token tile of one expert.
//    wave w: M-subtile mt = w>>2 (16 rows), column group cg = w&3.
// ---------------------------------------------------------------------------
__global__ __launch_bounds__(256)
void moe_tile_kernel(const float* __restrict__ x,
                     const float* __restrict__ pes,
                     const float* __restrict__ gw_all,
                     const float* __restrict__ uw_all,
                     const float* __restrict__ dw_all,
                     const int*   __restrict__ counts,
                     const int*   __restrict__ offsets,
                     const int*   __restrict__ bucket,
                     const float* __restrict__ topk_w,
                     float* __restrict__ out) {
    const int e    = blockIdx.x / MAX_TILES;
    const int tile = blockIdx.x % MAX_TILES;
    const int cnt  = counts[e];
    const int base = tile * TILE_M;
    if (base >= cnt) return;                        // uniform early-exit

    __shared__ __bf16 xs[TILE_M * XS_STRIDE];       // ~66 KB: gathered x rows (bf16)
    __shared__ __bf16 hs[TILE_M * HS_STRIDE];       // ~33 KB: activation chunk (bf16)
    __shared__ int    toks[TILE_M];
    __shared__ float  wts[TILE_M];

    const int tid = threadIdx.x;

    // row -> (token, routing weight); pad with weight 0 so EXEC stays all-1s
    if (tid < TILE_M) {
        int gi = base + tid;
        int tok = 0; float wt = 0.0f;
        if (gi < cnt) {
            int a = bucket[offsets[e] + gi];
            tok = a >> 1;
            wt  = topk_w[a];
        }
        toks[tid] = tok;
        wts[tid]  = wt;
    }
    __syncthreads();

    // gather 32 token rows of x into LDS, fp32 -> bf16
    {
        int r     = tid >> 3;          // 0..31 row
        int cbase = (tid & 7) * 128;   // 128 floats per thread
        const float* src = x + (size_t)toks[r] * EMBED + cbase;
        __bf16* dst = xs + r * XS_STRIDE + cbase;
#pragma unroll
        for (int i = 0; i < 32; ++i) {
            float4 v = ((const float4*)src)[i];
            dst[i * 4 + 0] = (__bf16)v.x;
            dst[i * 4 + 1] = (__bf16)v.y;
            dst[i * 4 + 2] = (__bf16)v.z;
            dst[i * 4 + 3] = (__bf16)v.w;
        }
    }
    __syncthreads();

    const int wave  = tid >> 5;
    const int lane  = tid & 31;
    const int mt    = wave >> 2;            // M-subtile (0/1)
    const int cg    = wave & 3;             // column group
    const int nl    = lane & 15;            // N column within tile / A row within subtile
    const int m1    = mt * 16 + nl;         // A-matrix row
    const int khalf = (lane >> 4) << 3;     // K half select: 0 or 8
    const int mhalf = (lane >> 4) << 3;     // C row half: 0 or 8

    const float* gw = gw_all + (size_t)e * EMBED  * EXPERT;
    const float* uw = uw_all + (size_t)e * EMBED  * EXPERT;
    const float* dw = dw_all + (size_t)e * EXPERT * EMBED;

    // persistent down-proj accumulators: wave owns 16 rows x 256 cols = 16 tiles
    v8f oacc[16];
#pragma unroll
    for (int j = 0; j < 16; ++j) oacc[j] = {};

#pragma unroll 1
    for (int c = 0; c < NCHUNK; ++c) {
        __syncthreads();   // protect hs reuse across chunks

        // ---- stage 1: h[32, CHUNK] = gelu(x*Wg) * (x*Wu) for this chunk ----
        // wave covers 128 expert-cols = 8 WMMA N-tiles for its 16-row subtile
#pragma unroll 1
        for (int j = 0; j < 8; ++j) {
            const int cn = c * CHUNK + cg * 128 + j * 16;    // global expert col
            v8f ag = {}, au = {};
#pragma unroll 1
            for (int k0 = 0; k0 < EMBED; k0 += 32) {
                const int kb = k0 + khalf;
                v16bf av, bg, bu;
#pragma unroll
                for (int i = 0; i < 8; ++i) {
                    av[i]     = xs[m1 * XS_STRIDE + kb + i];
                    av[i + 8] = xs[m1 * XS_STRIDE + kb + 16 + i];
                }
                const int ncol = cn + nl;
                // branch-free next-K prefetch (wraps at the end, harmless)
                const int kpf = ((k0 + 32) & (EMBED - 1)) + khalf;
                __builtin_prefetch(&gw[(size_t)kpf * EXPERT + ncol], 0, 1);
                __builtin_prefetch(&uw[(size_t)kpf * EXPERT + ncol], 0, 1);
#pragma unroll
                for (int i = 0; i < 8; ++i) {
                    const size_t k1 = (size_t)(kb + i)      * EXPERT + ncol;
                    const size_t k2 = (size_t)(kb + 16 + i) * EXPERT + ncol;
                    bg[i]     = (__bf16)gw[k1];
                    bg[i + 8] = (__bf16)gw[k2];
                    bu[i]     = (__bf16)uw[k1];
                    bu[i + 8] = (__bf16)uw[k2];
                }
                ag = __builtin_amdgcn_wmma_f32_16x16x32_bf16(false, av, false, bg,
                                                             (short)0, ag, false, false);
                au = __builtin_amdgcn_wmma_f32_16x16x32_bf16(false, av, false, bu,
                                                             (short)0, au, false, false);
            }
            // tanh-gelu(g)*u == g * sigmoid(1.59577*(g + 0.044715 g^3)) * u
            const int hcol = cg * 128 + j * 16 + nl;
#pragma unroll
            for (int r = 0; r < 8; ++r) {
                const int mr = mt * 16 + mhalf + r;
                float g = ag[r], u = au[r];
                float s  = 1.5957691216057308f * (g + 0.044715f * g * g * g);
                float gl = g * fast_sigmoid(s);              // exp + rcp, branch-free
                hs[mr * HS_STRIDE + hcol] = (__bf16)(gl * u);
            }
        }
        __syncthreads();

        // ---- stage 2: out += h_chunk * Wd[chunk rows, :] ----
#pragma unroll
        for (int j = 0; j < 16; ++j) {
            const int on = cg * 256 + j * 16;
            v8f acc = oacc[j];
#pragma unroll 1
            for (int k0 = 0; k0 < CHUNK; k0 += 32) {
                const int kb = k0 + khalf;
                v16bf av, bd;
#pragma unroll
                for (int i = 0; i < 8; ++i) {
                    av[i]     = hs[m1 * HS_STRIDE + kb + i];
                    av[i + 8] = hs[m1 * HS_STRIDE + kb + 16 + i];
                }
                const int ncol = on + nl;
                const int kpf  = c * CHUNK + ((k0 + 32) & (CHUNK - 1)) + khalf;
                __builtin_prefetch(&dw[(size_t)kpf * EMBED + ncol], 0, 1);
#pragma unroll
                for (int i = 0; i < 8; ++i) {
                    const size_t k1 = (size_t)(c * CHUNK + kb + i)      * EMBED + ncol;
                    const size_t k2 = (size_t)(c * CHUNK + kb + 16 + i) * EMBED + ncol;
                    bd[i]     = (__bf16)dw[k1];
                    bd[i + 8] = (__bf16)dw[k2];
                }
                acc = __builtin_amdgcn_wmma_f32_16x16x32_bf16(false, av, false, bd,
                                                              (short)0, acc, false, false);
            }
            oacc[j] = acc;
        }
    }

    // ---- epilogue: scale by per-expert scale * routing weight, atomic add ----
    const float es = pes[e];
#pragma unroll
    for (int j = 0; j < 16; ++j) {
        const int on = cg * 256 + j * 16 + nl;
#pragma unroll
        for (int r = 0; r < 8; ++r) {
            const int mr = mt * 16 + mhalf + r;
            float v = oacc[j][r] * es * wts[mr];
            atomicAdd(&out[(size_t)toks[mr] * EMBED + on], v);
        }
    }
}

// ---------------------------------------------------------------------------
// launcher
// ---------------------------------------------------------------------------
extern "C" void kernel_launch(void* const* d_in, const int* in_sizes, int n_in,
                              void* d_out, int out_size, void* d_ws, size_t ws_size,
                              hipStream_t stream) {
    (void)in_sizes; (void)n_in; (void)out_size; (void)ws_size;

    const float* x      = (const float*)d_in[0];
    const float* ginp   = (const float*)d_in[1];
    const float* gk     = (const float*)d_in[2];
    const float* pes    = (const float*)d_in[3];
    const float* gw_all = (const float*)d_in[4];
    const float* uw_all = (const float*)d_in[5];
    const float* dw_all = (const float*)d_in[6];
    float* out = (float*)d_out;

    // workspace layout (bytes)
    char* ws = (char*)d_ws;
    float* logits   = (float*)(ws + 0);                 // 2048*8*4   = 65536
    int*   topk_idx = (int*)  (ws + 65536);             // 4096*4     = 16384
    float* topk_w   = (float*)(ws + 65536 + 16384);     // 4096*4     = 16384
    int*   counts   = (int*)  (ws + 98304);             // 8*4
    int*   offsets  = (int*)  (ws + 98304 + 256);       // 8*4
    int*   cursors  = (int*)  (ws + 98304 + 512);       // 8*4
    int*   bucket   = (int*)  (ws + 98304 + 768);       // 4096*4     = 16384

    zero_kernel<<<(NTOK * EMBED + 255) / 256, 256, 0, stream>>>(out, counts);

    gate_logits_kernel<<<(NTOK * NUM_EXPERTS + 255) / 256, 256, 0, stream>>>(ginp, gk, logits);

    topk_kernel<<<(NTOK + 255) / 256, 256, 0, stream>>>(logits, topk_idx, topk_w, counts);

    scan_kernel<<<1, 64, 0, stream>>>(counts, offsets, cursors);

    scatter_kernel<<<(NASSIGN + 255) / 256, 256, 0, stream>>>(topk_idx, offsets, cursors, bucket);

    moe_tile_kernel<<<NUM_EXPERTS * MAX_TILES, 256, 0, stream>>>(
        x, pes, gw_all, uw_all, dw_all, counts, offsets, bucket, topk_w, out);
}
